// DistanceShareInefficientHeadProbEncoder_89446988907145
// MI455X (gfx1250) — matrix operand order repaired and checked by reading, not compiled, not run on hardware
//
#include <hip/hip_runtime.h>
#include <hip/hip_bf16.h>

// ---------------------------------------------------------------------------
// DistanceShareInefficientHeadProbEncoder on MI455X (gfx1250, wave32, WMMA)
//
// Pipeline per iteration (x4):
//   k_softmax : q_z = softmax(scores) -> Q bf16 [z][i][a] and Qt bf16 [z][a][i]
//   k_left    : left_k[z,c,i,b] = sum_a Q[z,i,a] * ternary[k,a,b,c]     (WMMA)
//   k_fsm     : TDM (tensor_load_to_lds) stages the [2][12][16][64] `left`
//               slab into LDS once per workgroup; 8 waves compute F tiles
//               (WMMA, A from LDS) + 12-head softmax in registers
//               -> qh bf16 [z][c][i][j]  AND  qhT bf16 [z][c][j][i]
//   k_msg x2  : t1_k = mask_k(qh) @ Q ; t2_k = mask'_k(qhT) @ Q         (WMMA)
//               triangular K-blocks skipped entirely; only the diagonal
//               block pays per-element masking
//   k_g       : scores = x + sum_k (t1_k @ Tab_k + t2_k @ Tswap_k)      (WMMA)
//
// Working set ~139 MB -> L2-resident on MI455X (192 MB L2), so materializing
// qh/qhT as bf16 costs L2 bandwidth, not HBM.
// ---------------------------------------------------------------------------

typedef __attribute__((ext_vector_type(16))) __bf16 v16bf;
typedef __attribute__((ext_vector_type(8)))  __bf16 v8bf;
typedef __attribute__((ext_vector_type(8)))  float  v8f;
typedef __attribute__((ext_vector_type(4)))  unsigned int v4u;
typedef __attribute__((ext_vector_type(8)))  int v8i;
typedef __attribute__((ext_vector_type(4)))  int v4i;

#define B_  8
#define L_  512
#define D_  64
#define H_  12
#define BC_ 768   // D_*H_

static __device__ __forceinline__ v8f vzero() {
  v8f z;
#pragma unroll
  for (int s = 0; s < 8; ++s) z[s] = 0.0f;
  return z;
}

// A-operand 16x32 bf16 tile from row-major memory [16 rows][stride>=32].
// Per ISA 7.12.2 (16-bit A 16x32): lane l: M=l&15, h=l>>4;
//   elems 0..7  <- K = h*8 .. h*8+7 ; elems 8..15 <- K = 16+h*8 .. 16+h*8+7
static __device__ __forceinline__ v16bf load_a_tile(const __hip_bfloat16* base, int stride) {
  int l = threadIdx.x & 31;
  int r = l & 15, h = l >> 4;
  const __bf16* p = (const __bf16*)base + (size_t)r * stride + h * 8;
  v8bf lo = *(const v8bf*)(p);
  v8bf hi = *(const v8bf*)(p + 16);
  return __builtin_shufflevector(lo, hi, 0,1,2,3,4,5,6,7,8,9,10,11,12,13,14,15);
}

// K index held by element e in half h of an A tile (used for diagonal masking).
static __device__ __forceinline__ int kmap(int e, int h) {
  return (e & 7) + 8 * h + 16 * (e >> 3);
}

// B-operand 32x16 bf16 tile; memory holds B^T row-major [16 N][stride>=32 K].
// lane l: N=l&15, h=l>>4; elems 0..7 <- K = h*8 .. h*8+7 ; elems 8..15 <- K = 16+h*8 .. 16+h*8+7.
static __device__ __forceinline__ v16bf load_b_tile(const __hip_bfloat16* baseT, int stride) {
  int l = threadIdx.x & 31;
  int n = l & 15, h = l >> 4;
  const __bf16* p = (const __bf16*)baseT + (size_t)n * stride + h * 8;
  v8bf lo = *(const v8bf*)(p);
  v8bf hi = *(const v8bf*)(p + 16);
  return __builtin_shufflevector(lo, hi, 0,1,2,3,4,5,6,7,8,9,10,11,12,13,14,15);
}

static __device__ __forceinline__ v8f wmma_bf16(v16bf a, v16bf b, v8f c) {
  return __builtin_amdgcn_wmma_f32_16x16x32_bf16(false, a, false, b, (short)0, c, false, false);
}

// ---------------------------------------------------------------------------
// Weight prep: bf16 copies of ternary in three orientations.
//   Tab [k][a][b*12+c] = ternary[k][a][b][c]   (B^T for G1 GEMM)
//   Tba [k][b*12+c][a] = ternary[k][a][b][c]   (B^T for left GEMM)
//   Tsw [k][b][a*12+c] = ternary[k][a][b][c]   (B^T for G2: swapped a<->b)
// ---------------------------------------------------------------------------
__global__ void k_prep(const float* __restrict__ tern,
                       __hip_bfloat16* __restrict__ Tab,
                       __hip_bfloat16* __restrict__ Tba,
                       __hip_bfloat16* __restrict__ Tsw) {
  int idx = blockIdx.x * blockDim.x + threadIdx.x;
  if (idx >= 2 * 64 * 64 * 12) return;
  int c = idx % 12;
  int b = (idx / 12) % 64;
  int a = (idx / 768) % 64;
  int k = idx / (768 * 64);
  __hip_bfloat16 bv = __float2bfloat16(tern[idx]);
  Tab[((size_t)k * 64 + a) * 768 + b * 12 + c] = bv;
  Tba[((size_t)k * 768 + b * 12 + c) * 64 + a] = bv;
  Tsw[((size_t)k * 64 + b) * 768 + a * 12 + c] = bv;
}

// ---------------------------------------------------------------------------
// Row softmax: one wave per row of 64. Emits Q (row-major) and Qt (transposed).
// ---------------------------------------------------------------------------
__global__ void k_softmax(const float* __restrict__ S, const int* __restrict__ mask,
                          __hip_bfloat16* __restrict__ Q, __hip_bfloat16* __restrict__ Qt) {
  int lane = threadIdx.x & 31;
  int row  = blockIdx.x * (blockDim.x >> 5) + (threadIdx.x >> 5);
  if (row >= B_ * L_) return;
  float x0 = S[(size_t)row * 64 + lane * 2];
  float x1 = S[(size_t)row * 64 + lane * 2 + 1];
  float m = fmaxf(x0, x1);
#pragma unroll
  for (int o = 16; o >= 1; o >>= 1) m = fmaxf(m, __shfl_xor(m, o, 32));
  float e0 = __expf(x0 - m), e1 = __expf(x1 - m);
  float s = e0 + e1;
#pragma unroll
  for (int o = 16; o >= 1; o >>= 1) s += __shfl_xor(s, o, 32);
  float inv = (mask[row] != 0) ? (1.0f / s) : 0.0f;
  __hip_bfloat16 q0 = __float2bfloat16(e0 * inv);
  __hip_bfloat16 q1 = __float2bfloat16(e1 * inv);
  Q[(size_t)row * 64 + lane * 2]     = q0;
  Q[(size_t)row * 64 + lane * 2 + 1] = q1;
  int z = row >> 9, i = row & 511;
  Qt[((size_t)z * 64 + lane * 2) * L_ + i]     = q0;
  Qt[((size_t)z * 64 + lane * 2 + 1) * L_ + i] = q1;
}

// ---------------------------------------------------------------------------
// left_k = Q @ ternary_k : C[i, bc] tiles. Output scattered to [k][z][c][i][b].
// grid: k*z*it*nt = 2*8*32*48, one wave per 16x16 tile, K=64 (2 WMMA).
// ---------------------------------------------------------------------------
__global__ void k_left(const __hip_bfloat16* __restrict__ Q,
                       const __hip_bfloat16* __restrict__ Tba,
                       __hip_bfloat16* __restrict__ left) {
  int t  = blockIdx.x;
  int nt = t % 48; t /= 48;
  int it = t % 32; t /= 32;
  int z  = t % 8;
  int k  = t / 8;
  int i0 = it * 16, n0 = nt * 16;
  const __hip_bfloat16* A  = Q   + ((size_t)z * L_ + i0) * 64;
  const __hip_bfloat16* Bt = Tba + ((size_t)k * BC_ + n0) * 64;
  v8f acc = vzero();
#pragma unroll
  for (int ks = 0; ks < 2; ++ks) {
    v16bf a = load_a_tile(A  + ks * 32, 64);
    v16bf b = load_b_tile(Bt + ks * 32, 64);
    acc = wmma_bf16(a, b, acc);
  }
  int l = threadIdx.x & 31, h = l >> 4;
  int n = n0 + (l & 15);
  int bcol = n / 12, c = n % 12;
#pragma unroll
  for (int r = 0; r < 8; ++r) {
    int i = i0 + r + 8 * h;
    left[(((size_t)(k * 8 + z) * H_ + c) * L_ + i) * 64 + bcol] = __float2bfloat16(acc[r]);
  }
}

// ---------------------------------------------------------------------------
// F tiles + head softmax -> qh [z][c][i][j] and qhT [z][c][j][i] (bf16).
// Workgroup = (z, i-strip, jt-quadrant): 8 waves, wave w -> jt = quad*8 + w.
// The `left` slab [2][12][16][64] (48 KB) is staged into LDS once per block
// via the Tensor Data Mover (tensor_load_to_lds, TENSORcnt), then all waves
// read A-tiles from LDS; global VMEM only feeds B-tiles.
// Triangle selection: jt>it -> k=0, jt<it -> k=1, jt==it -> per-element select.
// qhT store is a single packed b128 per head: each lane holds 8 consecutive i
// for its fixed j in the C/D register layout.
// ---------------------------------------------------------------------------
__global__ void __launch_bounds__(256) k_fsm(const __hip_bfloat16* __restrict__ left,
                                             const __hip_bfloat16* __restrict__ Q,
                                             __hip_bfloat16* __restrict__ qh,
                                             __hip_bfloat16* __restrict__ qhT) {
  __shared__ __hip_bfloat16 shL[2 * H_ * 16 * 64];  // [k][c][row][b], 48 KB

  int t    = blockIdx.x;
  int quad = t % 4;  t /= 4;
  int it   = t % 32;
  int z    = t / 32;
  int i0   = it * 16;

#if __has_builtin(__builtin_amdgcn_tensor_load_to_lds)
  if (threadIdx.x < 32) {  // wave 0 issues the two DMA descriptors
#pragma unroll
    for (int k = 0; k < 2; ++k) {
      // 3D tile: x = 64 elems (contig), y = 16 rows (stride 64),
      //          z = 12 head planes (stride 512*64); data_size = 2 bytes.
      unsigned long long ga =
          (unsigned long long)(size_t)(left + (((size_t)(k * 8 + z) * H_) * L_ + i0) * 64);
      unsigned lds = (unsigned)(size_t)(&shL[0]) + (unsigned)k * (H_ * 16 * 64 * 2);
      v4u g0 = { 1u,                                   // count=1, user descriptor
                 lds,                                  // lds_addr (bytes)
                 (unsigned)ga,                         // global_addr[31:0]
                 (unsigned)((ga >> 32) & 0x01FFFFFFu) | (2u << 30) };  // addr[56:32] | type=2
      v8i g1 = { (int)(1u << 16),                      // data_size=1 (2 bytes)
                 (int)(64u << 16),                     // tensor_dim0 = 64
                 (int)(512u << 16),                    // tensor_dim1 = 512
                 (int)(64u << 16),                     // tile_dim0 = 64
                 (int)(16u | (12u << 16)),             // tile_dim1=16, tile_dim2=12
                 64,                                   // tensor_dim0_stride = 64
                 (int)(32768u << 16),                  // tensor_dim1_stride = 32768 (lo16)
                 0 };
      v4i g2 = { H_, 0, 0, 0 };                        // tensor_dim2 = 12
      v4i g3 = { 0, 0, 0, 0 };
      v8i g4 = { 0, 0, 0, 0, 0, 0, 0, 0 };             // extra group (clang-23 form)
      __builtin_amdgcn_tensor_load_to_lds(g0, g1, g2, g3, g4, 0);
    }
    __builtin_amdgcn_s_wait_tensorcnt(0);
  }
#else
  // Fallback: cooperative vectorized copy of the slab into LDS.
  for (int idx = threadIdx.x; idx < 2 * H_ * 16 * 64 / 8; idx += 256) {
    int elem = idx * 8;
    int k    = elem / (H_ * 16 * 64);
    int rem  = elem % (H_ * 16 * 64);
    int c    = rem / (16 * 64);
    int r2   = rem % (16 * 64);
    int row  = r2 / 64, b = r2 % 64;
    const __hip_bfloat16* src =
        left + (((size_t)(k * 8 + z) * H_ + c) * L_ + i0 + row) * 64 + b;
    *(v8bf*)&shL[elem] = *(const v8bf*)src;
  }
#endif
  __syncthreads();

  int wv = threadIdx.x >> 5;
  int jt = quad * 8 + wv;
  int j0 = jt * 16;
  const __hip_bfloat16* Qz = Q + (size_t)z * L_ * 64;
  v16bf b0 = load_b_tile(Qz + (size_t)j0 * 64, 64);
  v16bf b1 = load_b_tile(Qz + (size_t)j0 * 64 + 32, 64);
  int l = threadIdx.x & 31, h = l >> 4, nn = l & 15;

  v8f acc[12];
  if (it != jt) {
    int k = (jt > it) ? 0 : 1;
#pragma unroll
    for (int c = 0; c < 12; ++c) {
      const __hip_bfloat16* Ab = &shL[(k * H_ + c) * 16 * 64];
      v8f a = vzero();
      a = wmma_bf16(load_a_tile(Ab, 64),      b0, a);
      a = wmma_bf16(load_a_tile(Ab + 32, 64), b1, a);
      acc[c] = a;
    }
  } else {
#pragma unroll
    for (int c = 0; c < 12; ++c) {
      const __hip_bfloat16* A0 = &shL[(0 * H_ + c) * 16 * 64];
      const __hip_bfloat16* A1 = &shL[(1 * H_ + c) * 16 * 64];
      v8f a0 = vzero(), a1 = vzero();
      a0 = wmma_bf16(load_a_tile(A0, 64),      b0, a0);
      a0 = wmma_bf16(load_a_tile(A0 + 32, 64), b1, a0);
      a1 = wmma_bf16(load_a_tile(A1, 64),      b0, a1);
      a1 = wmma_bf16(load_a_tile(A1 + 32, 64), b1, a1);
      v8f r;
#pragma unroll
      for (int s = 0; s < 8; ++s) {
        int mi = s + 8 * h;  // local i
        r[s] = (nn > mi) ? a0[s] : ((nn < mi) ? a1[s] : 0.0f);
      }
      acc[c] = r;
    }
  }

  // softmax over the 12 heads per (i,j) slot (REG_H == 1); overwrite acc with p
#pragma unroll
  for (int s = 0; s < 8; ++s) {
    float m = acc[0][s];
#pragma unroll
    for (int c = 1; c < 12; ++c) m = fmaxf(m, acc[c][s]);
    float sum = 0.0f;
    float e[12];
#pragma unroll
    for (int c = 0; c < 12; ++c) { e[c] = __expf(acc[c][s] - m); sum += e[c]; }
    float inv = 1.0f / sum;
#pragma unroll
    for (int c = 0; c < 12; ++c) acc[c][s] = e[c] * inv;
  }

  // qh[z][c][i][j]: 2B stores, lanes 0..15 cover consecutive j (coalesced)
#pragma unroll
  for (int s = 0; s < 8; ++s) {
    int i = i0 + s + 8 * h, j = j0 + nn;
#pragma unroll
    for (int c = 0; c < 12; ++c) {
      qh[(((size_t)z * H_ + c) * L_ + i) * L_ + j] = __float2bfloat16(acc[c][s]);
    }
  }
  // qhT[z][c][j][i]: one packed 16B store per head (8 consecutive i per lane)
#pragma unroll
  for (int c = 0; c < 12; ++c) {
    v8bf pk;
#pragma unroll
    for (int s = 0; s < 8; ++s) pk[s] = (__bf16)acc[c][s];
    __bf16* p = (__bf16*)qhT + (((size_t)z * H_ + c) * L_ + (j0 + nn)) * L_ + i0 + 8 * h;
    *(v8bf*)p = pk;
  }
}

// ---------------------------------------------------------------------------
// Message GEMM (covers both t1 and t2):
//   dst[k][z][i][b*12+c] = sum_j keep(i,j) * src[z][c][i][j] * Q[z][j][b]
// keep(i,j) = gt ? (j>i) : (j<i), with gt = (k==0) ^ invert.
//   t1: src=qh,  invert=0 ; t2: src=qhT, invert=1
// Triangular K-blocks: skipped / unmasked / per-element-masked (diagonal only).
// grid: k*z*c*it = 2*8*12*32, one wave per 16-row strip, N=64 (4 acc tiles).
// ---------------------------------------------------------------------------
__global__ void k_msg(const __hip_bfloat16* __restrict__ src,
                      const __hip_bfloat16* __restrict__ Qt,
                      __hip_bfloat16* __restrict__ dst, int invert) {
  int t  = blockIdx.x;
  int it = t % 32; t /= 32;
  int c  = t % 12; t /= 12;
  int z  = t % 8;
  int k  = t / 8;
  int gt = (k == 0) ^ invert;
  int i0 = it * 16;
  const __hip_bfloat16* M   = src + (((size_t)z * H_ + c) * L_ + i0) * L_;
  const __hip_bfloat16* QtZ = Qt + (size_t)z * 64 * L_;
  int l = threadIdx.x & 31, h = l >> 4;
  int i = i0 + (l & 15);
  v8f acc[4] = {vzero(), vzero(), vzero(), vzero()};
  for (int ks = 0; ks < 16; ++ks) {
    int jmin = ks * 32, jmax = jmin + 31;
    bool skip, full;
    if (gt) { skip = (jmax <= i0);      full = (jmin > i0 + 15); }
    else    { skip = (jmin >= i0 + 15); full = (jmax < i0);      }
    if (skip) continue;                      // uniform: no loads, no WMMA
    v16bf a = load_a_tile(M + jmin, L_);
    if (!full) {                             // only the diagonal block masks
#pragma unroll
      for (int e = 0; e < 16; ++e) {
        int j = jmin + kmap(e, h);
        bool keep = gt ? (j > i) : (j < i);
        a[e] = keep ? a[e] : (__bf16)0.0f;
      }
    }
#pragma unroll
    for (int nt = 0; nt < 4; ++nt) {
      v16bf b = load_b_tile(QtZ + (size_t)(nt * 16) * L_ + jmin, L_);
      acc[nt] = wmma_bf16(a, b, acc[nt]);
    }
  }
#pragma unroll
  for (int nt = 0; nt < 4; ++nt) {
    int bcol = nt * 16 + (l & 15);
#pragma unroll
    for (int r = 0; r < 8; ++r) {
      int ii = i0 + r + 8 * h;
      dst[((size_t)(k * 8 + z) * L_ + ii) * BC_ + bcol * 12 + c] = __float2bfloat16(acc[nt][r]);
    }
  }
}

// ---------------------------------------------------------------------------
// G: scores[z,i,a] = x[z,i,a] + sum_k ( t1_k @ Tab_k + t2_k @ Tswap_k )[i,a].
// grid: z*it*nt = 8*32*4, one wave per 16x16 output tile, K=768 per source.
// ---------------------------------------------------------------------------
__global__ void k_g(const __hip_bfloat16* __restrict__ t1f,
                    const __hip_bfloat16* __restrict__ t2f,
                    const __hip_bfloat16* __restrict__ Tab,
                    const __hip_bfloat16* __restrict__ Tsw,
                    const float* __restrict__ unary,
                    float* __restrict__ out) {
  int t  = blockIdx.x;
  int nt = t % 4; t /= 4;
  int it = t % 32;
  int z  = t / 32;
  int i0 = it * 16, n0 = nt * 16;
  v8f acc = vzero();
#pragma unroll
  for (int s = 0; s < 4; ++s) {
    int k = s & 1;
    const __hip_bfloat16* src = ((s < 2) ? t1f : t2f) + ((size_t)(k * 8 + z) * L_ + i0) * BC_;
    const __hip_bfloat16* W   = ((s < 2) ? Tab : Tsw) + ((size_t)k * 64 + n0) * BC_;
    for (int ks = 0; ks < 24; ++ks) {
      v16bf a = load_a_tile(src + ks * 32, BC_);
      v16bf b = load_b_tile(W   + ks * 32, BC_);
      acc = wmma_bf16(a, b, acc);
    }
  }
  int l = threadIdx.x & 31, h = l >> 4;
  int aa = n0 + (l & 15);
#pragma unroll
  for (int r = 0; r < 8; ++r) {
    int ii = i0 + r + 8 * h;
    size_t off = ((size_t)z * L_ + ii) * 64 + aa;
    out[off] = unary[off] + acc[r];
  }
}

// ---------------------------------------------------------------------------
// Host launcher: workspace carve + 4-iteration pipeline on `stream`.
// ---------------------------------------------------------------------------
extern "C" void kernel_launch(void* const* d_in, const int* in_sizes, int n_in,
                              void* d_out, int out_size, void* d_ws, size_t ws_size,
                              hipStream_t stream) {
  (void)in_sizes; (void)n_in; (void)out_size; (void)ws_size;
  const float* x       = (const float*)d_in[0];
  const int*   mask    = (const int*)d_in[1];
  const float* ternary = (const float*)d_in[2];
  float* out = (float*)d_out;
  char* ws = (char*)d_ws;

  // 256B-aligned workspace layout (total ~139 MB; L2-resident on MI455X)
  constexpr size_t OFF_Q      = 0;                      // bf16 [8][512][64]
  constexpr size_t OFF_QT     = OFF_Q      + 524288;    // bf16 [8][64][512]
  constexpr size_t OFF_TAB    = OFF_QT     + 524288;    // bf16 [2][64][768]
  constexpr size_t OFF_TBA    = OFF_TAB    + 196608;    // bf16 [2][768][64]
  constexpr size_t OFF_TSW    = OFF_TBA    + 196608;    // bf16 [2][64][768]
  constexpr size_t OFF_SCORES = OFF_TSW    + 196608;    // f32  [8][512][64]
  constexpr size_t OFF_LEFT   = OFF_SCORES + 1048576;   // bf16 [2][8][12][512][64]
  constexpr size_t OFF_QH     = OFF_LEFT   + 12582912;  // bf16 [8][12][512][512]
  constexpr size_t OFF_QHT    = OFF_QH     + 50331648;  // bf16 [8][12][512][512] (T)
  constexpr size_t OFF_T1     = OFF_QHT    + 50331648;  // bf16 [2][8][512][768]
  constexpr size_t OFF_T2     = OFF_T1     + 12582912;  // bf16 [2][8][512][768]

  __hip_bfloat16* Q    = (__hip_bfloat16*)(ws + OFF_Q);
  __hip_bfloat16* Qt   = (__hip_bfloat16*)(ws + OFF_QT);
  __hip_bfloat16* Tab  = (__hip_bfloat16*)(ws + OFF_TAB);
  __hip_bfloat16* Tba  = (__hip_bfloat16*)(ws + OFF_TBA);
  __hip_bfloat16* Tsw  = (__hip_bfloat16*)(ws + OFF_TSW);
  float*          Sc   = (float*)(ws + OFF_SCORES);
  __hip_bfloat16* left = (__hip_bfloat16*)(ws + OFF_LEFT);
  __hip_bfloat16* qh   = (__hip_bfloat16*)(ws + OFF_QH);
  __hip_bfloat16* qhT  = (__hip_bfloat16*)(ws + OFF_QHT);
  __hip_bfloat16* t1f  = (__hip_bfloat16*)(ws + OFF_T1);
  __hip_bfloat16* t2f  = (__hip_bfloat16*)(ws + OFF_T2);

  k_prep<<<384, 256, 0, stream>>>(ternary, Tab, Tba, Tsw);

  const float* S = x;
  for (int iter = 0; iter < 4; ++iter) {
    k_softmax<<<512, 256, 0, stream>>>(S, mask, Q, Qt);
    k_left<<<2 * 8 * 32 * 48, 32, 0, stream>>>(Q, Tba, left);
    k_fsm<<<8 * 32 * 4, 256, 0, stream>>>(left, Q, qh, qhT);
    k_msg<<<2 * 8 * 12 * 32, 32, 0, stream>>>(qh,  Qt, t1f, 0);
    k_msg<<<2 * 8 * 12 * 32, 32, 0, stream>>>(qhT, Qt, t2f, 1);
    float* dst = (iter < 3) ? Sc : out;
    k_g<<<8 * 32 * 4, 32, 0, stream>>>(t1f, t2f, Tab, Tsw, x, dst);
    S = dst;
  }
}
